// layer_self_attention_35038343201511
// MI455X (gfx1250) — matrix operand order repaired and checked by reading, not compiled, not run on hardware
//
#include <hip/hip_runtime.h>

typedef __attribute__((ext_vector_type(16))) __bf16 v16bf;
typedef __attribute__((ext_vector_type(8)))  float  v8f;

#define NHEAD 8
#define SPAT  784
#define TPAD  800
#define CIN   512
#define QC    256
#define KC    256
#define VC    256
#define DKH   32
#define BATCH 8

__device__ __forceinline__ unsigned short f32_to_bf16(float f) {
  unsigned u = __float_as_uint(f);
  u += 0x7FFFu + ((u >> 16) & 1u);          // round-to-nearest-even
  return (unsigned short)(u >> 16);
}

// ---------------------------------------------------------------------------
// Generic TN bf16 WMMA GEMM core (wave32).
//   A: M x K bf16, row-major (K contiguous), lda in uint32 units (=K/2)
//   B: N x K bf16, row-major (K contiguous), ldb in uint32 units
//   Computes C[m][n] += sum_k A[m][k]*B[n][k] for a 32x(16*NT) wave tile.
// Fragment layouts per CDNA5 ISA 7.12.2:
//   A lane: m = lane%16, K pairs at {0,2,4,6,16,18,20,22} + 8*(lane/16)
//   B lane: n = lane%16, K pairs at {0,2,4,6,16,18,20,22} + 8*(lane/16)
//   C lane: n = lane%16, VGPR j holds m = j + 8*(lane/16)
// ---------------------------------------------------------------------------
template<int MT, int NT>
__device__ __forceinline__ void gemm_tn(const unsigned* __restrict__ A, int ldaU,
                                        const unsigned* __restrict__ Bm, int ldbU,
                                        int M, int N, int K,
                                        int m0, int n0, int lane,
                                        v8f acc[MT][NT])
{
  const int half = lane >> 4;
  const int l16  = lane & 15;
  int ra[MT], rb[NT];
#pragma unroll
  for (int mt = 0; mt < MT; ++mt) { int r = m0 + mt*16 + l16; ra[mt] = (r < M) ? r : (M - 1); }
#pragma unroll
  for (int nt = 0; nt < NT; ++nt) { int r = n0 + nt*16 + l16; rb[nt] = (r < N) ? r : (N - 1); }

  for (int k0 = 0; k0 < K; k0 += 32) {
    const int kU = k0 >> 1;                 // uint32 (bf16-pair) offset
    v16bf afrag[MT], bfrag[NT];
#pragma unroll
    for (int mt = 0; mt < MT; ++mt) {
      union { unsigned u[8]; v16bf v; } t;
#pragma unroll
      for (int i = 0; i < 8; ++i) {
        const int off = ((i < 4) ? i : (i + 4)) + 4*half;   // {0..3,8..11}+4*half
        t.u[i] = A[(size_t)ra[mt]*ldaU + kU + off];
      }
      afrag[mt] = t.v;
    }
#pragma unroll
    for (int nt = 0; nt < NT; ++nt) {
      union { unsigned u[8]; v16bf v; } t;
#pragma unroll
      for (int i = 0; i < 8; ++i)
        t.u[i] = Bm[(size_t)rb[nt]*ldbU + kU + ((i < 4) ? i : (i + 4)) + 4*half]; // {0..3,8..11}+4*half
      bfrag[nt] = t.v;
    }
#pragma unroll
    for (int mt = 0; mt < MT; ++mt)
#pragma unroll
      for (int nt = 0; nt < NT; ++nt)
        acc[mt][nt] = __builtin_amdgcn_wmma_f32_16x16x32_bf16(
            false, afrag[mt], false, bfrag[nt], (short)0, acc[mt][nt], false, false);
  }
}

// ---------------------------------------------------------------------------
// Prep kernels
// ---------------------------------------------------------------------------
__global__ void __launch_bounds__(256) k_f32_to_bf16(const float* __restrict__ src,
                                                     unsigned short* __restrict__ dst, int n) {
  int i = blockIdx.x * 256 + threadIdx.x;
  if (i < n) dst[i] = f32_to_bf16(src[i]);
}

// X[b][c][s] f32  ->  Xt[b][s][c] bf16   (32x32 LDS tile transpose)
__global__ void __launch_bounds__(256) k_transpose(const float* __restrict__ X,
                                                   unsigned short* __restrict__ Xt,
                                                   int C, int S) {
  __shared__ float tile[32][33];
  const int tx = threadIdx.x & 31, ty = threadIdx.x >> 5;   // 32 x 8
  const int s0 = blockIdx.x * 32, c0 = blockIdx.y * 32, b = blockIdx.z;
  const float* Xb = X + (size_t)b * C * S;
#pragma unroll
  for (int r = 0; r < 4; ++r) {
    int c = c0 + ty + r*8, s = s0 + tx;
    if (s < S) tile[ty + r*8][tx] = Xb[(size_t)c * S + s];
  }
  __syncthreads();
#pragma unroll
  for (int r = 0; r < 4; ++r) {
    int s = s0 + ty + r*8, c = c0 + tx;
    if (s < S) Xt[((size_t)b * S + s) * C + c] = f32_to_bf16(tile[tx][ty + r*8]);
  }
}

// zero the K-padding stripe [padStart, padStart+padCount) of each row
__global__ void __launch_bounds__(256) k_zero_pad(unsigned short* __restrict__ buf,
                                                  int rowStride, int padStart, int padCount,
                                                  int nRows) {
  int i = blockIdx.x * 256 + threadIdx.x;
  if (i < nRows * padCount) {
    int r = i / padCount, t = i % padCount;
    buf[(size_t)r * rowStride + padStart + t] = 0;
  }
}

// ---------------------------------------------------------------------------
// K1a: Q projection.  C[o][s] = Wq[o][:]·Xt[s][:] + bq[o], scaled by 32^-0.5,
// stored transposed per head: qt[b][nh][s][d] bf16.
// ---------------------------------------------------------------------------
__global__ void __launch_bounds__(256) k_proj_q(const unsigned* __restrict__ Wb,
                                                const unsigned* __restrict__ Xt,
                                                const float* __restrict__ bias,
                                                unsigned short* __restrict__ qt) {
  const int lane = threadIdx.x & 31, wave = threadIdx.x >> 5;
  const int wm = wave & 3, wn = wave >> 2;           // 4x2 waves, 128x128 block
  const int m0 = blockIdx.x * 128 + wm * 32;
  const int n0 = blockIdx.y * 128 + wn * 64;
  const int b  = blockIdx.z;
  const unsigned* Bp = Xt + (size_t)b * SPAT * (CIN/2);
  v8f acc[2][4] = {};
  gemm_tn<2,4>(Wb, CIN/2, Bp, CIN/2, QC, SPAT, CIN, m0, n0, lane, acc);
  const int half = lane >> 4, l16 = lane & 15;
#pragma unroll
  for (int mt = 0; mt < 2; ++mt)
#pragma unroll
    for (int nt = 0; nt < 4; ++nt)
#pragma unroll
      for (int j = 0; j < 8; ++j) {
        int m = m0 + mt*16 + j + 8*half;
        int n = n0 + nt*16 + l16;
        if (m < QC && n < SPAT) {
          float v = (acc[mt][nt][j] + bias[m]) * 0.17677669529663687f; // 32^-0.5
          int nh = m >> 5, d = m & 31;
          qt[(((size_t)b*NHEAD + nh)*SPAT + n)*DKH + d] = f32_to_bf16(v);
        }
      }
}

// ---------------------------------------------------------------------------
// K1b: KV projection.  o<256 -> kt[b][nh][t][d];  o>=256 -> vm[b][nh][d][TPAD]
// ---------------------------------------------------------------------------
__global__ void __launch_bounds__(256) k_proj_kv(const unsigned* __restrict__ Wb,
                                                 const unsigned* __restrict__ Xt,
                                                 const float* __restrict__ bias,
                                                 unsigned short* __restrict__ kt,
                                                 unsigned short* __restrict__ vm) {
  const int lane = threadIdx.x & 31, wave = threadIdx.x >> 5;
  const int wm = wave & 3, wn = wave >> 2;
  const int m0 = blockIdx.x * 128 + wm * 32;
  const int n0 = blockIdx.y * 128 + wn * 64;
  const int b  = blockIdx.z;
  const unsigned* Bp = Xt + (size_t)b * SPAT * (CIN/2);
  v8f acc[2][4] = {};
  gemm_tn<2,4>(Wb, CIN/2, Bp, CIN/2, KC+VC, SPAT, CIN, m0, n0, lane, acc);
  const int half = lane >> 4, l16 = lane & 15;
#pragma unroll
  for (int mt = 0; mt < 2; ++mt)
#pragma unroll
    for (int nt = 0; nt < 4; ++nt)
#pragma unroll
      for (int j = 0; j < 8; ++j) {
        int m = m0 + mt*16 + j + 8*half;
        int n = n0 + nt*16 + l16;
        if (m < KC+VC && n < SPAT) {
          float v = acc[mt][nt][j] + bias[m];
          if (m < KC) {
            int nh = m >> 5, d = m & 31;
            kt[(((size_t)b*NHEAD + nh)*SPAT + n)*DKH + d] = f32_to_bf16(v);
          } else {
            int o = m - KC, nh = o >> 5, d = o & 31;
            vm[(((size_t)b*NHEAD + nh)*DKH + d)*TPAD + n] = f32_to_bf16(v);
          }
        }
      }
}

// ---------------------------------------------------------------------------
// K2: fused logits + softmax-over-heads.
// For a 128x64 (s,t) block tile, compute logits[nh][s][t] for all 8 heads
// (one K=32 WMMA group per head), keep all 8 in registers, softmax across nh,
// write weights bf16 to wgt[b][nh][s][TPAD].  Eliminates the 157MB logits
// round trip to HBM.
// ---------------------------------------------------------------------------
__global__ void __launch_bounds__(256) k_logits_softmax(const unsigned* __restrict__ qt,
                                                        const unsigned* __restrict__ kt,
                                                        unsigned short* __restrict__ wgt) {
  const int lane = threadIdx.x & 31, wave = threadIdx.x >> 5;
  const int wm = wave & 3, wn = wave >> 2;           // 4x2 waves, 128x64 block
  const int m0 = blockIdx.x * 128 + wm * 32;
  const int n0 = blockIdx.y * 64  + wn * 32;
  const int b  = blockIdx.z;
  v8f acc[NHEAD][2][2] = {};
#pragma unroll
  for (int nh = 0; nh < NHEAD; ++nh) {
    const unsigned* qh = qt + (size_t)(b*NHEAD + nh) * SPAT * (DKH/2);
    const unsigned* kh = kt + (size_t)(b*NHEAD + nh) * SPAT * (DKH/2);
    gemm_tn<2,2>(qh, DKH/2, kh, DKH/2, SPAT, SPAT, DKH, m0, n0, lane, acc[nh]);
  }
  const int half = lane >> 4, l16 = lane & 15;
#pragma unroll
  for (int mt = 0; mt < 2; ++mt)
#pragma unroll
    for (int nt = 0; nt < 2; ++nt)
#pragma unroll
      for (int j = 0; j < 8; ++j) {
        int m = m0 + mt*16 + j + 8*half;
        int n = n0 + nt*16 + l16;
        if (m < SPAT && n < SPAT) {
          float mx = acc[0][mt][nt][j];
#pragma unroll
          for (int nh = 1; nh < NHEAD; ++nh) mx = fmaxf(mx, acc[nh][mt][nt][j]);
          float e[NHEAD], s = 0.f;
#pragma unroll
          for (int nh = 0; nh < NHEAD; ++nh) { e[nh] = __expf(acc[nh][mt][nt][j] - mx); s += e[nh]; }
          float inv = 1.f / s;
#pragma unroll
          for (int nh = 0; nh < NHEAD; ++nh)
            wgt[(((size_t)b*NHEAD + nh)*SPAT + m)*TPAD + n] = f32_to_bf16(e[nh] * inv);
        }
      }
}

// ---------------------------------------------------------------------------
// K3: attention x V.  attn[s][d] = sum_t wgt[s][t]*v[d][t], K=TPAD (zero-padded).
// Scatter-store implementing the reference's raw reshape (s*32+d -> c*784+p),
// transposed to XoT[b][p][ch] bf16 so the out-projection is a TN GEMM.
// ---------------------------------------------------------------------------
__global__ void __launch_bounds__(256) k_attn(const unsigned* __restrict__ wgt,
                                              const unsigned* __restrict__ vm,
                                              unsigned short* __restrict__ xot) {
  const int lane = threadIdx.x & 31, wave = threadIdx.x >> 5;
  const int m0 = blockIdx.x * 256 + wave * 32;       // 8 waves along M, 256x32 block
  const int z  = blockIdx.z;                         // b*8 + nh
  const int b  = z >> 3, nh = z & 7;
  const unsigned* A  = wgt + (size_t)z * SPAT * (TPAD/2);
  const unsigned* Bp = vm  + (size_t)z * DKH  * (TPAD/2);
  v8f acc[2][2] = {};
  gemm_tn<2,2>(A, TPAD/2, Bp, TPAD/2, SPAT, DKH, TPAD, m0, 0, lane, acc);
  const int half = lane >> 4, l16 = lane & 15;
#pragma unroll
  for (int mt = 0; mt < 2; ++mt)
#pragma unroll
    for (int nt = 0; nt < 2; ++nt)
#pragma unroll
      for (int j = 0; j < 8; ++j) {
        int s = m0 + mt*16 + j + 8*half;
        int d = nt*16 + l16;
        if (s < SPAT) {
          int i  = s * DKH + d;                      // reference reshape scramble
          int cl = i / SPAT, p = i % SPAT;
          int ch = nh * DKH + cl;
          xot[((size_t)b * SPAT + p) * VC + ch] = f32_to_bf16(acc[mt][nt][j]);
        }
      }
}

// ---------------------------------------------------------------------------
// K4: output projection.  out[b][o][p] = Wo[o][:]·XoT[b][p][:] + bo[o]  (f32 out)
// ---------------------------------------------------------------------------
__global__ void __launch_bounds__(256) k_outproj(const unsigned* __restrict__ Wb,
                                                 const unsigned* __restrict__ xot,
                                                 const float* __restrict__ bias,
                                                 float* __restrict__ out) {
  const int lane = threadIdx.x & 31, wave = threadIdx.x >> 5;
  const int wm = wave & 3, wn = wave >> 2;
  const int m0 = blockIdx.x * 128 + wm * 32;
  const int n0 = blockIdx.y * 128 + wn * 64;
  const int b  = blockIdx.z;
  const unsigned* Bp = xot + (size_t)b * SPAT * (VC/2);
  v8f acc[2][4] = {};
  gemm_tn<2,4>(Wb, VC/2, Bp, VC/2, CIN, SPAT, VC, m0, n0, lane, acc);
  const int half = lane >> 4, l16 = lane & 15;
#pragma unroll
  for (int mt = 0; mt < 2; ++mt)
#pragma unroll
    for (int nt = 0; nt < 4; ++nt)
#pragma unroll
      for (int j = 0; j < 8; ++j) {
        int m = m0 + mt*16 + j + 8*half;
        int n = n0 + nt*16 + l16;
        if (m < CIN && n < SPAT)
          out[((size_t)b * CIN + m) * SPAT + n] = acc[mt][nt][j] + bias[m];
      }
}

// ---------------------------------------------------------------------------
extern "C" void kernel_launch(void* const* d_in, const int* in_sizes, int n_in,
                              void* d_out, int out_size, void* d_ws, size_t ws_size,
                              hipStream_t stream) {
  (void)in_sizes; (void)n_in; (void)out_size; (void)ws_size;
  const float* inputs    = (const float*)d_in[0];
  const float* inputs_st = (const float*)d_in[1];
  const float* Wq  = (const float*)d_in[2];
  const float* bq  = (const float*)d_in[3];
  const float* Wkv = (const float*)d_in[4];
  const float* bkv = (const float*)d_in[5];
  const float* Wo  = (const float*)d_in[6];
  const float* bo  = (const float*)d_in[7];
  float* out = (float*)d_out;

  char* ws = (char*)d_ws;
  size_t off = 0;
  auto carve = [&](size_t elems) -> unsigned short* {
    unsigned short* p = (unsigned short*)(ws + off);
    off += ((elems * 2 + 255) & ~(size_t)255);
    return p;
  };
  unsigned short* XtQ  = carve((size_t)BATCH * SPAT * CIN);       // [b][s][c] bf16
  unsigned short* XtKV = carve((size_t)BATCH * SPAT * CIN);
  unsigned short* WqB  = carve((size_t)QC * CIN);
  unsigned short* WkvB = carve((size_t)(KC + VC) * CIN);
  unsigned short* WoB  = carve((size_t)CIN * VC);
  unsigned short* Qt   = carve((size_t)BATCH * NHEAD * SPAT * DKH); // [b][nh][s][d]
  unsigned short* Kt   = carve((size_t)BATCH * NHEAD * SPAT * DKH); // [b][nh][t][d]
  unsigned short* Vm   = carve((size_t)BATCH * NHEAD * DKH * TPAD); // [b][nh][d][tpad]
  unsigned short* Wgt  = carve((size_t)BATCH * NHEAD * SPAT * TPAD);// [b][nh][s][tpad]
  unsigned short* XoT  = carve((size_t)BATCH * SPAT * VC);          // [b][p][ch]

  // prep: weights to bf16, activations transposed+converted, K-pads zeroed
  k_f32_to_bf16<<<dim3((QC*CIN + 255)/256), 256, 0, stream>>>(Wq, WqB, QC*CIN);
  k_f32_to_bf16<<<dim3(((KC+VC)*CIN + 255)/256), 256, 0, stream>>>(Wkv, WkvB, (KC+VC)*CIN);
  k_f32_to_bf16<<<dim3((CIN*VC + 255)/256), 256, 0, stream>>>(Wo, WoB, CIN*VC);
  k_transpose<<<dim3(25, CIN/32, BATCH), 256, 0, stream>>>(inputs,    XtQ,  CIN, SPAT);
  k_transpose<<<dim3(25, CIN/32, BATCH), 256, 0, stream>>>(inputs_st, XtKV, CIN, SPAT);
  k_zero_pad<<<dim3((BATCH*NHEAD*DKH*16 + 255)/256), 256, 0, stream>>>(Vm,  TPAD, SPAT, 16, BATCH*NHEAD*DKH);
  k_zero_pad<<<dim3((BATCH*NHEAD*SPAT*16 + 255)/256), 256, 0, stream>>>(Wgt, TPAD, SPAT, 16, BATCH*NHEAD*SPAT);

  // pipeline
  k_proj_q <<<dim3(2, 7, BATCH), 256, 0, stream>>>((const unsigned*)WqB,  (const unsigned*)XtQ,  bq,  Qt);
  k_proj_kv<<<dim3(4, 7, BATCH), 256, 0, stream>>>((const unsigned*)WkvB, (const unsigned*)XtKV, bkv, Kt, Vm);
  k_logits_softmax<<<dim3(7, 13, BATCH), 256, 0, stream>>>((const unsigned*)Qt, (const unsigned*)Kt, Wgt);
  k_attn   <<<dim3(4, 1, BATCH*NHEAD), 256, 0, stream>>>((const unsigned*)Wgt, (const unsigned*)Vm, XoT);
  k_outproj<<<dim3(4, 7, BATCH), 256, 0, stream>>>((const unsigned*)WoB, (const unsigned*)XoT, bo, out);
}
